// GATEncoder_20495583936894
// MI455X (gfx1250) — matrix operand order repaired and checked by reading, not compiled, not run on hardware
//
#include <hip/hip_runtime.h>
#include <hip/hip_bf16.h>

// ---------------------------------------------------------------------------
// GAT encoder (3x GATConv + VAE dense head) for MI455X / gfx1250, wave32.
// GEMMs: v_wmma_f32_16x16x32_f16, 16x64 strip per wave, guard-free b128 loads.
// Edge softmax: ordered-uint atomicMax + f32 atomicAdd (L2-resident).
// ---------------------------------------------------------------------------

typedef __attribute__((ext_vector_type(16))) _Float16 v16h;
typedef __attribute__((ext_vector_type(8)))  float    v8f;

static constexpr int Bsz = 256, Nn = 128, BN = Bsz * Nn, NE = 524288;
static constexpr int F = 8, H = 4, H1 = 32, H2 = 16, H3 = 512, LAT = 128, C = 32;

// ---- ordered-uint encoding so atomicMax(u32) == float max ------------------
__device__ __forceinline__ unsigned f2ord(float f) {
    unsigned u = __float_as_uint(f);
    return (u & 0x80000000u) ? ~u : (u | 0x80000000u);
}
__device__ __forceinline__ float ord2f(unsigned u) {
    unsigned b = (u & 0x80000000u) ? (u & 0x7FFFFFFFu) : ~u;
    return __uint_as_float(b);
}
#define NEG_INF_ORD 0x007FFFFFu  // f2ord(-inf)

__device__ __forceinline__ float4 relu4(float4 v) {
    v.x = fmaxf(v.x, 0.0f); v.y = fmaxf(v.y, 0.0f);
    v.z = fmaxf(v.z, 0.0f); v.w = fmaxf(v.w, 0.0f);
    return v;
}

// pack two float4 pairs into low/high 8 halves of a v16h fragment
__device__ __forceinline__ void cvt8(v16h& f, int base, float4 lo, float4 hi) {
    f[base + 0] = (_Float16)lo.x; f[base + 1] = (_Float16)lo.y;
    f[base + 2] = (_Float16)lo.z; f[base + 3] = (_Float16)lo.w;
    f[base + 4] = (_Float16)hi.x; f[base + 5] = (_Float16)hi.y;
    f[base + 6] = (_Float16)hi.z; f[base + 7] = (_Float16)hi.w;
}

// B fragment (contiguous k order, not the ISA table order): lane owns col, K = e + half*16 -> 16
// contiguous floats at Wr[kb .. kb+15].
__device__ __forceinline__ v16h loadB(const float* __restrict__ Wr, int kb) {
    float4 b0 = *(const float4*)(Wr + kb);
    float4 b1 = *(const float4*)(Wr + kb + 4);
    float4 b2 = *(const float4*)(Wr + kb + 8);
    float4 b3 = *(const float4*)(Wr + kb + 12);
    v16h b;
    cvt8(b, 0, b0, b1);
    cvt8(b, 8, b2, b3);
    return b;
}

// ---------------------------------------------------------------------------
// Generic GEMM: out[M,N] = act(A[M,K] * W[N,K]^T + bias) (+ addsrc)
// Requires M%64==0, N%64==0, K%32==0. One wave per 16(M)x64(N) strip.
// ---------------------------------------------------------------------------
__global__ __launch_bounds__(128)
void gemm_wmma_f16(const float* __restrict__ A, const float* __restrict__ W,
                   const float* __restrict__ bias, const float* __restrict__ addsrc,
                   float* __restrict__ out, int M, int N, int K,
                   int relu_in, int relu_out)
{
    const int wave = threadIdx.x >> 5;
    const int lane = threadIdx.x & 31;
    const int half = lane >> 4;          // 0: lanes 0-15, 1: lanes 16-31
    const int l16  = lane & 15;
    const int tm   = blockIdx.y * 4 + wave;
    const int col0 = blockIdx.x * 64 + l16;

    const float* Arow = A + (size_t)(tm * 16 + l16) * K + half * 16;
    const float* Wr0  = W + (size_t)(col0 +  0) * K + half * 16;
    const float* Wr1  = W + (size_t)(col0 + 16) * K + half * 16;
    const float* Wr2  = W + (size_t)(col0 + 32) * K + half * 16;
    const float* Wr3  = W + (size_t)(col0 + 48) * K + half * 16;

    v8f acc[4] = {v8f{}, v8f{}, v8f{}, v8f{}};

    for (int kb = 0; kb < K; kb += 32) {
        // A fragment (contiguous k order, not the ISA table order): K = kb + half*16 + e,
        // one contiguous 16-float run (matches B).
        float4 a0 = *(const float4*)(Arow + kb);
        float4 a1 = *(const float4*)(Arow + kb + 4);
        float4 a2 = *(const float4*)(Arow + kb + 8);
        float4 a3 = *(const float4*)(Arow + kb + 12);
        if (relu_in) { a0 = relu4(a0); a1 = relu4(a1); a2 = relu4(a2); a3 = relu4(a3); }
        v16h a;
        cvt8(a, 0, a0, a1);
        cvt8(a, 8, a2, a3);

        v16h b0 = loadB(Wr0, kb);
        v16h b1 = loadB(Wr1, kb);
        v16h b2 = loadB(Wr2, kb);
        v16h b3 = loadB(Wr3, kb);

        acc[0] = __builtin_amdgcn_wmma_f32_16x16x32_f16(false, a, false, b0, (short)0, acc[0], false, false);
        acc[1] = __builtin_amdgcn_wmma_f32_16x16x32_f16(false, a, false, b1, (short)0, acc[1], false, false);
        acc[2] = __builtin_amdgcn_wmma_f32_16x16x32_f16(false, a, false, b2, (short)0, acc[2], false, false);
        acc[3] = __builtin_amdgcn_wmma_f32_16x16x32_f16(false, a, false, b3, (short)0, acc[3], false, false);
    }

#pragma unroll
    for (int j = 0; j < 4; ++j) {
        const int col = col0 + j * 16;
        const float bval = bias ? bias[col] : 0.0f;
#pragma unroll
        for (int r = 0; r < 8; ++r) {
            const int orow = tm * 16 + r + half * 8;   // C/D layout: M = r + half*8
            float v = acc[j][r] + bval;
            if (relu_out) v = fmaxf(v, 0.0f);
            if (addsrc) v += addsrc[(size_t)orow * N + col];
            out[(size_t)orow * N + col] = v;
        }
    }
}

// ---- zero-pad K dimension (rows x Kin -> rows x Kout) ----------------------
__global__ void pad_k(const float* __restrict__ src, float* __restrict__ dst,
                      int rows, int Kin, int Kout)
{
    long long id = (long long)blockIdx.x * blockDim.x + threadIdx.x;
    if (id >= (long long)rows * Kout) return;
    int r = (int)(id / Kout), k = (int)(id % Kout);
    dst[id] = (k < Kin) ? src[(size_t)r * Kin + k] : 0.0f;
}

// ---------------------------------------------------------------------------
// Per-node attention logits: asrc[n,h] = <h[n,h,:], a_src[h,:]>, same for adst.
// ---------------------------------------------------------------------------
__global__ void attn_dots(const float* __restrict__ hbuf,
                          const float* __restrict__ a_src,
                          const float* __restrict__ a_dst,
                          float* __restrict__ asrc, float* __restrict__ adst, int HL)
{
    int id = blockIdx.x * blockDim.x + threadIdx.x;
    if (id >= BN * H) return;
    int n = id / H, hh = id % H;
    int Cout = H * HL;
    const float* hp = hbuf + (size_t)n * Cout + hh * HL;
    const float* as = a_src + hh * HL;
    const float* ad = a_dst + hh * HL;
    float s = 0.0f, d = 0.0f;
    for (int c = 0; c < HL; ++c) { float v = hp[c]; s += v * as[c]; d += v * ad[c]; }
    asrc[id] = s; adst[id] = d;
}

// ---- edge pass 1: e = leaky_relu(asrc[src]+adst[dst]); segment max over dst
__global__ void edge_score(const int* __restrict__ Es, const int* __restrict__ Ed,
                           const float* __restrict__ asrc, const float* __restrict__ adst,
                           float* __restrict__ ebuf, unsigned* __restrict__ mbuf)
{
    int id = blockIdx.x * blockDim.x + threadIdx.x;
    if (id >= NE * H) return;
    int e = id / H, hh = id % H;
    int s = Es[e], d = Ed[e];
    float v = asrc[s * H + hh] + adst[d * H + hh];
    v = (v > 0.0f) ? v : 0.2f * v;           // leaky_relu(0.2)
    ebuf[id] = v;
    atomicMax(&mbuf[d * H + hh], f2ord(v));
}

// ---- edge pass 2: ex = exp(e - m[dst]); segment sum of ex over dst
__global__ void edge_exp(const int* __restrict__ Ed, float* __restrict__ ebuf,
                         const unsigned* __restrict__ mbuf, float* __restrict__ den)
{
    int id = blockIdx.x * blockDim.x + threadIdx.x;
    if (id >= NE * H) return;
    int e = id / H, hh = id % H;
    int d = Ed[e];
    float m = ord2f(mbuf[d * H + hh]);
    if (!(m > -3.0e38f && m < 3.0e38f)) m = 0.0f;   // nodes w/o in-edges
    float ex = __expf(ebuf[id] - m);
    ebuf[id] = ex;
    atomicAdd(&den[d * H + hh], ex);
}

// ---- edge pass 3: out[dst] += h[src] * ex/(den[dst]+eps), 4 channels/thread
__global__ void edge_aggr(const int* __restrict__ Es, const int* __restrict__ Ed,
                          const float* __restrict__ ebuf, const float* __restrict__ den,
                          const float* __restrict__ hbuf, float* __restrict__ out, int HL)
{
    int chunks = HL / 4;
    int per = H * chunks;
    long long id = (long long)blockIdx.x * blockDim.x + threadIdx.x;
    if (id >= (long long)NE * per) return;
    int e  = (int)(id / per);
    int r  = (int)(id % per);
    int hh = r / chunks;
    int c4 = (r % chunks) * 4;
    int s = Es[e], d = Ed[e];
    float alpha = ebuf[e * H + hh] / (den[d * H + hh] + 1e-16f);
    int Cout = H * HL;
    const float4 hv = *(const float4*)(hbuf + (size_t)s * Cout + hh * HL + c4);
    float* op = out + (size_t)d * Cout + hh * HL + c4;
    atomicAdd(op + 0, hv.x * alpha);
    atomicAdd(op + 1, hv.y * alpha);
    atomicAdd(op + 2, hv.z * alpha);
    atomicAdd(op + 3, hv.w * alpha);
}

// ---- GAT3 head-mean + relu: x3[n,c] = relu(mean_h out3[n,h,c]) --------------
__global__ void head_mean_relu(const float* __restrict__ out3, float* __restrict__ x3)
{
    int id = blockIdx.x * blockDim.x + threadIdx.x;
    if (id >= BN * H2) return;
    int n = id / H2, c = id % H2;
    float s = 0.0f;
    for (int hh = 0; hh < H; ++hh) s += out3[(size_t)n * (H * H2) + hh * H2 + c];
    x3[id] = fmaxf(s * (1.0f / H), 0.0f);
}

// ---- boxes/labels small dense: mixin = relu(boxes@Wb^T+bb)+relu(labels@Wl^T+bl)
__global__ void box_label(const float* __restrict__ X, const float* __restrict__ Wb,
                          const float* __restrict__ bb, const float* __restrict__ Wl,
                          const float* __restrict__ bl, float* __restrict__ mixin)
{
    int id = blockIdx.x * blockDim.x + threadIdx.x;
    if (id >= BN * H2) return;
    int n = id / H2, c = id % H2;
    const float* xr = X + (size_t)n * F;
    float bsum = bb[c];
    for (int i = 0; i < F - 1; ++i) bsum += xr[1 + i] * Wb[c * (F - 1) + i];
    float lsum = bl[c] + xr[0] * Wl[c];
    mixin[id] = fmaxf(bsum, 0.0f) + fmaxf(lsum, 0.0f);
}

// ---- build [cl | x] concat rows ([B, C+2048]) ------------------------------
__global__ void build_xcat(const float* __restrict__ cl, const float* __restrict__ x3,
                           float* __restrict__ xcat)
{
    const int KW = C + Nn * H2;  // 2080
    int id = blockIdx.x * blockDim.x + threadIdx.x;
    if (id >= Bsz * KW) return;
    int b = id / KW, j = id % KW;
    xcat[id] = (j < C) ? cl[b * C + j] : x3[(size_t)b * (Nn * H2) + (j - C)];
}

// ---- fills -----------------------------------------------------------------
__global__ void fill_f32(float* p, size_t n, float v)
{ size_t i = (size_t)blockIdx.x * blockDim.x + threadIdx.x; if (i < n) p[i] = v; }
__global__ void fill_u32(unsigned* p, size_t n, unsigned v)
{ size_t i = (size_t)blockIdx.x * blockDim.x + threadIdx.x; if (i < n) p[i] = v; }

// ---------------------------------------------------------------------------
extern "C" void kernel_launch(void* const* d_in, const int* in_sizes, int n_in,
                              void* d_out, int out_size, void* d_ws, size_t ws_size,
                              hipStream_t stream)
{
    (void)in_sizes; (void)n_in; (void)out_size; (void)ws_size;

    const int*   E    = (const int*)d_in[0];
    const float* X    = (const float*)d_in[1];
    const float* cl   = (const float*)d_in[2];
    const float* g1W  = (const float*)d_in[3];
    const float* g1as = (const float*)d_in[4];
    const float* g1ad = (const float*)d_in[5];
    const float* g2W  = (const float*)d_in[6];
    const float* g2as = (const float*)d_in[7];
    const float* g2ad = (const float*)d_in[8];
    const float* g3W  = (const float*)d_in[9];
    const float* g3as = (const float*)d_in[10];
    const float* g3ad = (const float*)d_in[11];
    const float* Wb   = (const float*)d_in[12];
    const float* bb   = (const float*)d_in[13];
    const float* Wl   = (const float*)d_in[14];
    const float* bl   = (const float*)d_in[15];
    const float* W1   = (const float*)d_in[16];
    const float* b1   = (const float*)d_in[17];
    const float* W2   = (const float*)d_in[18];
    const float* b2   = (const float*)d_in[19];
    const float* W3   = (const float*)d_in[20];
    const float* b3   = (const float*)d_in[21];
    const float* Wm   = (const float*)d_in[22];
    const float* bm   = (const float*)d_in[23];
    const float* Wv   = (const float*)d_in[24];
    const float* bv   = (const float*)d_in[25];

    const int* Es = E;
    const int* Ed = E + NE;

    // ---- workspace carve-up (floats, 256B aligned) -- total ~91 MB ---------
    float* ws = (float*)d_ws;
    size_t off = 0;
    auto alloc = [&](size_t n) { float* p = ws + off; off += (n + 63) & ~(size_t)63; return p; };
    float*    h1    = alloc((size_t)BN * 128);
    float*    out1  = alloc((size_t)BN * 128);
    float*    h2    = alloc((size_t)BN * 64);
    float*    out2  = alloc((size_t)BN * 64);
    float*    h3    = alloc((size_t)BN * 64);
    float*    out3  = alloc((size_t)BN * 64);
    float*    x3    = alloc((size_t)BN * H2);
    float*    Xp    = alloc((size_t)BN * 32);     // X zero-padded K=8 -> 32
    float*    g1Wp  = alloc((size_t)128 * 32);    // gat1_W zero-padded
    float*    asrc  = alloc((size_t)BN * H);
    float*    adst  = alloc((size_t)BN * H);
    unsigned* mbuf  = (unsigned*)alloc((size_t)BN * H);
    float*    den   = alloc((size_t)BN * H);
    float*    ebuf  = alloc((size_t)NE * H);
    float*    mixin = alloc((size_t)Bsz * Nn * H2);
    float*    mix   = alloc((size_t)Bsz * H3);
    float*    xcat  = alloc((size_t)Bsz * (C + Nn * H2));
    float*    t1    = alloc((size_t)Bsz * H3);
    float*    t2    = alloc((size_t)Bsz * H3);
    float*    t3    = alloc((size_t)Bsz * H3);

    auto blocks = [](long long n) { return (unsigned)((n + 255) / 256); };
    auto gemm = [&](const float* A, const float* W, const float* bias, const float* add,
                    float* o, int M, int N, int K, int rin, int rout) {
        gemm_wmma_f16<<<dim3(N / 64, M / 64), 128, 0, stream>>>(A, W, bias, add, o,
                                                                M, N, K, rin, rout);
    };

    // ---- one GATConv layer -------------------------------------------------
    auto run_gat = [&](const float* xin, int Cin, int HL, const float* Wg,
                       const float* a_s, const float* a_d,
                       float* hbuf, float* obuf, int relu_in) {
        const int Cout = H * HL;
        fill_f32<<<blocks((long long)BN * Cout), 256, 0, stream>>>(obuf, (size_t)BN * Cout, 0.0f);
        fill_f32<<<blocks(BN * H), 256, 0, stream>>>(den, (size_t)BN * H, 0.0f);
        fill_u32<<<blocks(BN * H), 256, 0, stream>>>(mbuf, (size_t)BN * H, NEG_INF_ORD);
        gemm(xin, Wg, nullptr, nullptr, hbuf, BN, Cout, Cin, relu_in, 0);  // h = x@W^T
        attn_dots<<<blocks(BN * H), 256, 0, stream>>>(hbuf, a_s, a_d, asrc, adst, HL);
        edge_score<<<blocks((long long)NE * H), 256, 0, stream>>>(Es, Ed, asrc, adst, ebuf, mbuf);
        edge_exp<<<blocks((long long)NE * H), 256, 0, stream>>>(Ed, ebuf, mbuf, den);
        edge_aggr<<<blocks((long long)NE * H * (HL / 4)), 256, 0, stream>>>(
            Es, Ed, ebuf, den, hbuf, obuf, HL);
    };

    // ---- GAT stack (relu fused into next layer's A-load) -------------------
    pad_k<<<blocks((long long)BN * 32), 256, 0, stream>>>(X,   Xp,   BN,  F, 32);
    pad_k<<<blocks((long long)128 * 32), 256, 0, stream>>>(g1W, g1Wp, 128, F, 32);
    run_gat(Xp,   32,     H1, g1Wp, g1as, g1ad, h1, out1, 0);
    run_gat(out1, H * H1, H2, g2W,  g2as, g2ad, h2, out2, 1);
    run_gat(out2, H * H2, H2, g3W,  g3as, g3ad, h3, out3, 1);
    head_mean_relu<<<blocks(BN * H2), 256, 0, stream>>>(out3, x3);  // [B, 2048] contig

    // ---- dense head --------------------------------------------------------
    box_label<<<blocks(BN * H2), 256, 0, stream>>>(X, Wb, bb, Wl, bl, mixin);
    gemm(mixin, W1, b1, nullptr, mix, Bsz, H3, Nn * H2, 0, 1);           // mix
    build_xcat<<<blocks((long long)Bsz * (C + Nn * H2)), 256, 0, stream>>>(cl, x3, xcat);
    gemm(xcat, W2, b2, mix, t1, Bsz, H3, C + Nn * H2, 0, 1);             // relu(.)+mix
    gemm(t1, W3, b3, nullptr, t2, Bsz, H3, H3, 0, 1);                    // dense3 #1
    gemm(t2, W3, b3, nullptr, t3, Bsz, H3, H3, 0, 1);                    // dense3 #2 (shared W)
    float* zout = (float*)d_out;
    gemm(t3, Wm, bm, nullptr, zout,              Bsz, LAT, H3, 0, 0);    // z_mean
    gemm(t3, Wv, bv, nullptr, zout + Bsz * LAT,  Bsz, LAT, H3, 0, 0);    // z_logvar
}
